// BellmanLoss_77429670412386
// MI455X (gfx1250) — matrix operand order repaired and checked by reading, not compiled, not run on hardware
//
#include <hip/hip_runtime.h>

typedef __attribute__((ext_vector_type(16))) _Float16 v16h;
typedef __attribute__((ext_vector_type(8)))  _Float16 v8h;
typedef __attribute__((ext_vector_type(8)))  float    v8f;

#define HSTRIDE 264   // halves per LDS row (256 + 8 pad -> 4-bank step, 16B aligned)
#define WAVES   4
#define MROWS   32    // rows per wave (2 row tiles of 16)
#define BATCH   65536
#define SDIM    128
#define HDIM    256
#define ADIM    18
#define DONE_V  666.0f
#define GAMMA   0.9f

#define WMMA_F16(a, b, c) \
  __builtin_amdgcn_wmma_f32_16x16x32_f16(false, (a), false, (b), (short)0, (c), false, false)

// ---------------- weight prep: fp32 row-major -> f16 transposed [N][K] -------
extern "C" __global__ void bellman_prep(const float* __restrict__ W1,
                                        const float* __restrict__ W2,
                                        const float* __restrict__ W3,
                                        _Float16* __restrict__ W1t,
                                        _Float16* __restrict__ W2t,
                                        _Float16* __restrict__ W3t) {
  int i = blockIdx.x * blockDim.x + threadIdx.x;
  if (i < 256 * 128) {                       // W1t[n][k] = W1[k][n], K=128
    int n = i >> 7, k = i & 127;
    W1t[i] = (_Float16)W1[k * 256 + n];
    return;
  }
  int j = i - 256 * 128;
  if (j >= 0 && j < 256 * 256) {             // W2t[n][k] = W2[k][n], K=256
    int n = j >> 8, k = j & 255;
    W2t[j] = (_Float16)W2[k * 256 + n];
    return;
  }
  int l = i - 256 * 128 - 256 * 256;
  if (l >= 0 && l < 32 * 256) {              // W3t[n][k], n padded 18->32 with zeros
    int n = l >> 8, k = l & 255;
    W3t[l] = (_Float16)((n < ADIM) ? W3[k * ADIM + n] : 0.0f);
  }
}

// -------- fragment loaders (contiguous k = 16h + j order for A and B, wave32) ----------------------
// A 16x32 f16: lane(h=l>>4, m=l&15): halves[j] = K(h*16 + j), j=0..15
__device__ __forceinline__ v16h load_afrag(const _Float16* buf, int kt, int lane) {
  int h = (lane >> 4) & 1, m = lane & 15;
  const _Float16* p = buf + m * HSTRIDE + kt * 32 + h * 16;
  v8h lo = *(const v8h*)(p);
  v8h hi = *(const v8h*)(p + 8);
  return __builtin_shufflevector(lo, hi, 0,1,2,3,4,5,6,7,8,9,10,11,12,13,14,15);
}

// B 32x16 f16: lane(h,n): halves[j] = W[K = kt*32 + h*16 + j][col nt*16+n]
// Wt stored transposed [N][K] -> contiguous 32B per lane.
__device__ __forceinline__ v16h load_bfrag(const _Float16* Wt, int K, int nt, int kt, int lane) {
  int h = (lane >> 4) & 1, n = lane & 15;
  return *(const v16h*)(Wt + (size_t)(nt * 16 + n) * K + kt * 32 + h * 16);
}

// -------- store one f32 C/D tile as f16 with bias/relu -----------------------
__device__ __forceinline__ void store_tile(_Float16* outbuf, const v8f& acc, float bv,
                                           bool relu, int nt, int lane) {
  const int h = (lane >> 4) & 1, n = lane & 15;
#pragma unroll
  for (int r = 0; r < 8; ++r) {
    float v = acc[r] + bv;
    if (relu) v = fmaxf(v, 0.0f);
    outbuf[(h * 8 + r) * HSTRIDE + nt * 16 + n] = (_Float16)v;   // C/D row = h*8+r
  }
}

// -------- dense layer, 2x2 register blocking: out[32][Nout] ------------------
// Each B fragment feeds 2 row tiles; each A fragment feeds 2 col tiles:
// ~1 KB of fragment traffic per 2 WMMAs instead of per 1.
template <int K, int NT, bool RELU>
__device__ __forceinline__ void layer(const _Float16* inbuf, _Float16* outbuf,
                                      const _Float16* Wt, const float* bias, int lane) {
  const int n = lane & 15;
  const _Float16* in0 = inbuf;
  const _Float16* in1 = inbuf + 16 * HSTRIDE;
  _Float16* out0 = outbuf;
  _Float16* out1 = outbuf + 16 * HSTRIDE;
  for (int nt = 0; nt < NT; nt += 2) {
    v8f a00 = {0.f,0.f,0.f,0.f,0.f,0.f,0.f,0.f};
    v8f a01 = a00, a10 = a00, a11 = a00;
#pragma unroll
    for (int kt = 0; kt < K / 32; ++kt) {
      v16h af0 = load_afrag(in0, kt, lane);
      v16h af1 = load_afrag(in1, kt, lane);
      v16h bf0 = load_bfrag(Wt, K, nt,     kt, lane);
      v16h bf1 = load_bfrag(Wt, K, nt + 1, kt, lane);
      a00 = WMMA_F16(af0, bf0, a00);
      a01 = WMMA_F16(af0, bf1, a01);
      a10 = WMMA_F16(af1, bf0, a10);
      a11 = WMMA_F16(af1, bf1, a11);
    }
    float bv0 = bias[nt * 16 + n];
    float bv1 = bias[(nt + 1) * 16 + n];
    store_tile(out0, a00, bv0, RELU, nt,     lane);
    store_tile(out0, a01, bv1, RELU, nt + 1, lane);
    store_tile(out1, a10, bv0, RELU, nt,     lane);
    store_tile(out1, a11, bv1, RELU, nt + 1, lane);
  }
  asm volatile("s_wait_dscnt 0x0" ::: "memory");
}

// -------- final layer -> f32 Q [32 rows][32 cols] in LDS ---------------------
__device__ __forceinline__ void layer3(const _Float16* inbuf, float* Qbuf,
                                       const _Float16* W3t, const float* b3, int lane) {
  const int h = (lane >> 4) & 1, n = lane & 15;
  const _Float16* in0 = inbuf;
  const _Float16* in1 = inbuf + 16 * HSTRIDE;
  v8f a00 = {0.f,0.f,0.f,0.f,0.f,0.f,0.f,0.f};
  v8f a01 = a00, a10 = a00, a11 = a00;
#pragma unroll
  for (int kt = 0; kt < HDIM / 32; ++kt) {
    v16h af0 = load_afrag(in0, kt, lane);
    v16h af1 = load_afrag(in1, kt, lane);
    v16h bf0 = load_bfrag(W3t, HDIM, 0, kt, lane);
    v16h bf1 = load_bfrag(W3t, HDIM, 1, kt, lane);
    a00 = WMMA_F16(af0, bf0, a00);
    a01 = WMMA_F16(af0, bf1, a01);
    a10 = WMMA_F16(af1, bf0, a10);
    a11 = WMMA_F16(af1, bf1, a11);
  }
  float bv0 = b3[n];                                  // cols 0..15 always valid
  float bv1 = ((16 + n) < ADIM) ? b3[16 + n] : 0.0f;  // cols 16..31 padded
#pragma unroll
  for (int r = 0; r < 8; ++r) {
    int row0 = h * 8 + r, row1 = 16 + row0;
    Qbuf[row0 * 32 + n]      = a00[r] + bv0;
    Qbuf[row1 * 32 + n]      = a10[r] + bv0;
    float q01 = ((16 + n) < ADIM) ? (a01[r] + bv1) : -3.0e38f;  // pad cols lose max
    float q11 = ((16 + n) < ADIM) ? (a11[r] + bv1) : -3.0e38f;
    Qbuf[row0 * 32 + 16 + n] = q01;
    Qbuf[row1 * 32 + 16 + n] = q11;
  }
  asm volatile("s_wait_dscnt 0x0" ::: "memory");
}

// -------- stage 32 rows of fp32 states into f16 LDS (coalesced) --------------
__device__ __forceinline__ void stage(const float* __restrict__ X, int row_base,
                                      _Float16* buf, int lane) {
#pragma unroll 4
  for (int i = lane; i < MROWS * SDIM; i += 32) {
    int r = i >> 7, c = i & (SDIM - 1);
    buf[r * HSTRIDE + c] = (_Float16)X[(size_t)(row_base + r) * SDIM + c];
  }
  asm volatile("s_wait_dscnt 0x0" ::: "memory");
}

__device__ __forceinline__ void mlp32(const float* __restrict__ X, int row_base,
                                      _Float16* buf0, _Float16* buf1, float* Qbuf,
                                      const _Float16* W1t, const _Float16* W2t,
                                      const _Float16* W3t,
                                      const float* b1, const float* b2, const float* b3,
                                      int lane) {
  stage(X, row_base, buf0, lane);
  layer<SDIM, HDIM / 16, true>(buf0, buf1, W1t, b1, lane);   // 128 -> 256
  layer<HDIM, HDIM / 16, true>(buf1, buf0, W2t, b2, lane);   // 256 -> 256
  layer3(buf0, Qbuf, W3t, b3, lane);                         // 256 -> 18 (pad 32)
}

// ---------------------------- main kernel ------------------------------------
extern "C" __global__ void __launch_bounds__(WAVES * 32)
bellman_main(const float* __restrict__ states0, const int* __restrict__ actions,
             const float* __restrict__ rewards, const float* __restrict__ states1,
             const float* __restrict__ b1, const float* __restrict__ b2,
             const float* __restrict__ b3,
             const _Float16* __restrict__ W1t, const _Float16* __restrict__ W2t,
             const _Float16* __restrict__ W3t, float* __restrict__ partials) {
  extern __shared__ _Float16 smem[];
  __shared__ float wsum[WAVES];

  const int wave = threadIdx.x >> 5;
  const int lane = threadIdx.x & 31;
  _Float16* buf0 = smem + (size_t)wave * (2 * MROWS * HSTRIDE);
  _Float16* buf1 = buf0 + MROWS * HSTRIDE;
  float* Qbuf = (float*)buf1;   // 32x32 f32 Q tile overlays buf1 (free during layer3)

  const int row_base = blockIdx.x * (WAVES * MROWS) + wave * MROWS;
  const int row = row_base + lane;                 // one row per lane for epilogue

  // ---- pass 1: Qn(states1) -> Bellman target per row ----
  mlp32(states1, row_base, buf0, buf1, Qbuf, W1t, W2t, W3t, b1, b2, b3, lane);
  float mx = -3.0e38f;
#pragma unroll
  for (int c = 0; c < ADIM; ++c) mx = fmaxf(mx, Qbuf[lane * 32 + c]);
  float rw  = rewards[row];
  float s10 = states1[(size_t)row * SDIM];
  float tgt = (s10 == DONE_V) ? rw : rw + GAMMA * mx;
  asm volatile("s_wait_dscnt 0x0" ::: "memory");   // Qbuf consumed before reuse

  // ---- pass 2: Q0(states0) -> gather at argmax(actions) ----
  mlp32(states0, row_base, buf0, buf1, Qbuf, W1t, W2t, W3t, b1, b2, b3, lane);
  const int* arow = actions + (size_t)row * ADIM;
  int best = 0, bv = arow[0];
#pragma unroll
  for (int c = 1; c < ADIM; ++c) {
    int v = arow[c];
    if (v > bv) { bv = v; best = c; }
  }
  float d = Qbuf[lane * 32 + best] - tgt;
  float sq = d * d;

  // deterministic reduce: shuffle tree within wave, fixed order across waves
#pragma unroll
  for (int off = 16; off >= 1; off >>= 1) sq += __shfl_down(sq, off, 32);
  if (lane == 0) wsum[wave] = sq;
  __syncthreads();
  if (threadIdx.x == 0) {
    float s = 0.0f;
    for (int w = 0; w < WAVES; ++w) s += wsum[w];
    partials[blockIdx.x] = s;
  }
}

// -------- final fixed-order sum of per-block partials ------------------------
extern "C" __global__ void bellman_final(const float* __restrict__ partials,
                                         float* __restrict__ out, int n) {
  if (blockIdx.x == 0 && threadIdx.x == 0) {
    float s = 0.0f;
    for (int i = 0; i < n; ++i) s += partials[i];
    out[0] = s;
  }
}

// ---------------------------- host launcher ----------------------------------
extern "C" void kernel_launch(void* const* d_in, const int* in_sizes, int n_in,
                              void* d_out, int out_size, void* d_ws, size_t ws_size,
                              hipStream_t stream) {
  (void)in_sizes; (void)n_in; (void)out_size; (void)ws_size;
  const float* states0 = (const float*)d_in[0];
  const int*   actions = (const int*)d_in[1];
  const float* rewards = (const float*)d_in[2];
  const float* states1 = (const float*)d_in[3];
  const float* W1 = (const float*)d_in[4];
  const float* b1 = (const float*)d_in[5];
  const float* W2 = (const float*)d_in[6];
  const float* b2 = (const float*)d_in[7];
  const float* W3 = (const float*)d_in[8];
  const float* b3 = (const float*)d_in[9];

  char* ws = (char*)d_ws;
  _Float16* W1t = (_Float16*)(ws);                          // 256*128*2 = 64 KB
  _Float16* W2t = (_Float16*)(ws + 65536);                  // 256*256*2 = 128 KB
  _Float16* W3t = (_Float16*)(ws + 65536 + 131072);         // 32*256*2  = 16 KB
  float* partials = (float*)(ws + 65536 + 131072 + 16384);  // 512 floats

  const int prep_total = 256 * 128 + 256 * 256 + 32 * 256;
  bellman_prep<<<(prep_total + 255) / 256, 256, 0, stream>>>(W1, W2, W3, W1t, W2t, W3t);

  const int blocks = BATCH / (WAVES * MROWS);               // 512
  const size_t shmem = (size_t)WAVES * 2 * MROWS * HSTRIDE * sizeof(_Float16);  // 135,168 B
  hipFuncSetAttribute((const void*)bellman_main,
                      hipFuncAttributeMaxDynamicSharedMemorySize, (int)shmem);
  bellman_main<<<blocks, WAVES * 32, shmem, stream>>>(
      states0, actions, rewards, states1, b1, b2, b3, W1t, W2t, W3t, partials);

  bellman_final<<<1, 32, 0, stream>>>(partials, (float*)d_out, blocks);
}
